// HTP_59588376265255
// MI455X (gfx1250) — matrix operand order repaired and checked by reading, not compiled, not run on hardware
//
#include <hip/hip_runtime.h>

// ---------------------------------------------------------------------------
// MI455X (gfx1250) forward pass, live-path only (final_intent is provably
// uniform 1/T -> attention scores / mu_w / sigma_w / time_int are dead).
// Dense transforms: v_wmma_f32_16x16x32_f16, f16 operands pre-packed in
// memory so fragments load as aligned b128 vector loads (no per-lane cvt
// storms, no register spills).
// ---------------------------------------------------------------------------

#define HID    64
#define BSZ    256
#define TLEN   200
#define NU     6000
#define NI     10000
#define NC     400
#define N_UI   (NU + NI)    // 16000
#define N_UC   (NU + NC)    // 6400
#define NNZ_UI 512000
#define NNZ_UC 96000
#define NNZ_IT 30000

typedef __attribute__((ext_vector_type(16))) _Float16 v16h;
typedef __attribute__((ext_vector_type(8)))  _Float16 v8h;
typedef __attribute__((ext_vector_type(8)))  float    v8f;

static __device__ __forceinline__ v8f wmma16(v16h a, v16h b, v8f c) {
  return __builtin_amdgcn_wmma_f32_16x16x32_f16(false, a, false, b, (short)0, c,
                                                false, false);
}

// A fragment (16x32) from f16 row-major Xh (ld=64): lane m=lane&15, hi=lane>>4;
// elems 0..15 -> K=ks+hi*16+i (two b128).
static __device__ __forceinline__ v16h load_a_h(const _Float16* __restrict__ Xh,
                                                int r0, int ks) {
  int lane = threadIdx.x & 31;
  int m = lane & 15;
  int kb = ks + ((lane >> 4) << 4);
  const _Float16* p = Xh + (size_t)(r0 + m) * HID + kb;
  v8h lo = *(const v8h*)p;
  v8h hi = *(const v8h*)(p + 8);
  v16h a;
#pragma unroll
  for (int i = 0; i < 8; ++i) { a[i] = lo[i]; a[8 + i] = hi[i]; }
  return a;
}

// B fragment (32x16) for W^T, W f16 row-major (Nout x 64): lane n=lane&15,
// hi=lane>>4; 16 consecutive halfs at K=ks+hi*16 (32B aligned -> 2x b128).
static __device__ __forceinline__ v16h load_bT_h(const _Float16* __restrict__ Wh,
                                                 int n0, int ks) {
  int lane = threadIdx.x & 31;
  int n = lane & 15;
  int kb = ks + ((lane >> 4) << 4);
  return *(const v16h*)(Wh + (size_t)(n0 + n) * HID + kb);
}

static __device__ __forceinline__ float reduce_add_32(float v) {
#pragma unroll
  for (int off = 16; off; off >>= 1) v += __shfl_xor(v, off, 32);
  return v;
}
static __device__ __forceinline__ float reduce_add_16(float v) {
#pragma unroll
  for (int off = 8; off; off >>= 1) v += __shfl_xor(v, off, 32);  // stays in 16-group
  return v;
}

// ---------------------------------------------------------------------------
// utility kernels
// ---------------------------------------------------------------------------
__global__ void k_zero(float* p, int n) {
  int i = blockIdx.x * blockDim.x + threadIdx.x;
  if (i < n) p[i] = 0.f;
}
__global__ void k_scale(float* p, int n, float s) {
  int i = blockIdx.x * blockDim.x + threadIdx.x;
  if (i < n) p[i] *= s;
}
__global__ void k_cvt16(const float* __restrict__ src, _Float16* __restrict__ dst, int n) {
  int i = blockIdx.x * blockDim.x + threadIdx.x;
  if (i < n) dst[i] = (_Float16)src[i];
}
// side_h = f16(side); prod_h = f16(ego * side)   (A operands for NGCF WMMA)
__global__ void k_prep_ab(const float* __restrict__ side, const float* __restrict__ ego,
                          _Float16* __restrict__ side_h, _Float16* __restrict__ prod_h, int n) {
  int i = blockIdx.x * blockDim.x + threadIdx.x;
  if (i >= n) return;
  float s = side[i];
  side_h[i] = (_Float16)s;
  prod_h[i] = (_Float16)(ego[i] * s);
}
__global__ void k_init2(const float* a, int na, const float* b, float* ego, float* acc, int tot) {
  int i = blockIdx.x * blockDim.x + threadIdx.x;
  if (i >= tot) return;
  float v = (i < na) ? a[i] : b[i - na];
  ego[i] = v;
  acc[i] = v;
}
__global__ void k_concat3(const float* a, int na, const float* b, int nb,
                          const float* c, float* dst, int tot) {
  int i = blockIdx.x * blockDim.x + threadIdx.x;
  if (i >= tot) return;
  float v;
  if (i < na)           v = a[i];
  else if (i < na + nb) v = b[i - na];
  else                  v = c[i - na - nb];
  dst[i] = v;
}

// ---------------------------------------------------------------------------
// SpMM scatter: out[rows[e],:] += vals[e]*x[cols[e],:]  (1 wave/edge, grid-
// stride, prefetch next edge's gathered row -> global_prefetch path)
// ---------------------------------------------------------------------------
__global__ void k_spmm(const int* __restrict__ rows, const int* __restrict__ cols,
                       const float* __restrict__ vals, const float* __restrict__ x,
                       float* __restrict__ out, int nnz) {
  int nwaves = (gridDim.x * blockDim.x) >> 5;
  int wid = (blockIdx.x * blockDim.x + threadIdx.x) >> 5;
  int lane = threadIdx.x & 31;
  for (int e = wid; e < nnz; e += nwaves) {
    int en = e + nwaves;
    if (en < nnz) {
      int cn = cols[en];
      __builtin_prefetch(x + (size_t)cn * HID + lane, 0, 0);
    }
    int r = rows[e], c = cols[e];
    float v = vals[e];
    const float* xr = x + (size_t)c * HID;
    float* orow = out + (size_t)r * HID;
    atomicAdd(&orow[lane], v * xr[lane]);
    atomicAdd(&orow[lane + 32], v * xr[lane + 32]);
  }
}

// ---------------------------------------------------------------------------
// NGCF dense layer (WMMA, f16 operands):
//   s = lrelu(side@Wgc^T+bgc); b = lrelu((ego*side)@Wbi^T+bbi)
//   ego_out = s+b;  acc += ego_out * rsqrt(rowsum(ego_out^2)+1e-12)
// Per-wave 16-row tile; ego_out written per column tile (no e[4][8] held ->
// no spills), then an L0-hot reload pass applies the row-norm to acc.
// ---------------------------------------------------------------------------
__global__ void k_ngcf_dense(const _Float16* __restrict__ side_h,
                             const _Float16* __restrict__ prod_h,
                             const _Float16* __restrict__ Wgc_h, const float* __restrict__ bgc,
                             const _Float16* __restrict__ Wbi_h, const float* __restrict__ bbi,
                             float* __restrict__ ego_out, float* __restrict__ acc, int N) {
  int tile = blockIdx.x * 4 + (threadIdx.x >> 5);
  int r0 = tile * 16;
  if (r0 >= N) return;                 // wave-uniform
  int lane = threadIdx.x & 31;
  int hi = lane >> 4, nl = lane & 15;

  v16h a1_0 = load_a_h(side_h, r0, 0);
  v16h a1_1 = load_a_h(side_h, r0, 32);
  v16h a2_0 = load_a_h(prod_h, r0, 0);
  v16h a2_1 = load_a_h(prod_h, r0, 32);

  float rs[8];
#pragma unroll
  for (int r = 0; r < 8; ++r) rs[r] = 0.f;

#pragma unroll
  for (int t = 0; t < 4; ++t) {
    v8f c1 = {}, c2 = {};
    c1 = wmma16(a1_0, load_bT_h(Wgc_h, t * 16, 0), c1);
    c1 = wmma16(a1_1, load_bT_h(Wgc_h, t * 16, 32), c1);
    c2 = wmma16(a2_0, load_bT_h(Wbi_h, t * 16, 0), c2);
    c2 = wmma16(a2_1, load_bT_h(Wbi_h, t * 16, 32), c2);
    float bg = bgc[t * 16 + nl];
    float bb = bbi[t * 16 + nl];
#pragma unroll
    for (int r = 0; r < 8; ++r) {
      float s = c1[r] + bg;  s = (s > 0.f) ? s : 0.01f * s;
      float bv = c2[r] + bb; bv = (bv > 0.f) ? bv : 0.01f * bv;
      float ev = s + bv;
      ego_out[(size_t)(r0 + hi * 8 + r) * HID + t * 16 + nl] = ev;
      rs[r] += ev * ev;
    }
  }
  // full row L2 norm (e^2 summed across the 16 lanes of each half-group)
#pragma unroll
  for (int r = 0; r < 8; ++r) rs[r] = rsqrtf(reduce_add_16(rs[r]) + 1e-12f);

  // acc += ego_out * rn   (reload is L0-hot; same-wave store->load is ordered)
#pragma unroll
  for (int t = 0; t < 4; ++t) {
#pragma unroll
    for (int r = 0; r < 8; ++r) {
      size_t off = (size_t)(r0 + hi * 8 + r) * HID + t * 16 + nl;
      acc[off] += ego_out[off] * rs[r];
    }
  }
}

// ---------------------------------------------------------------------------
// xp = seqs(51200x64) @ wih^T(64x192) + bih  (WMMA, f16 operands)
// ---------------------------------------------------------------------------
__global__ void k_xp_gemm(const _Float16* __restrict__ Xh, const _Float16* __restrict__ Wih_h,
                          const float* __restrict__ bih, float* __restrict__ XP) {
  int tile = blockIdx.x * 4 + (threadIdx.x >> 5);
  int r0 = tile * 16;                   // 3200 tiles, always full
  int lane = threadIdx.x & 31;
  int hi = lane >> 4, nl = lane & 15;
  v16h a0 = load_a_h(Xh, r0, 0);
  v16h a1 = load_a_h(Xh, r0, 32);
#pragma unroll
  for (int nt = 0; nt < 12; ++nt) {
    v8f c = {};
    c = wmma16(a0, load_bT_h(Wih_h, nt * 16, 0), c);
    c = wmma16(a1, load_bT_h(Wih_h, nt * 16, 32), c);
    int col = nt * 16 + nl;
    float bb = bih[col];
#pragma unroll
    for (int r = 0; r < 8; ++r)
      XP[(size_t)(r0 + hi * 8 + r) * (3 * HID) + col] = c[r] + bb;
  }
}

// ---------------------------------------------------------------------------
// seqs[b,t,:] = (log_seqs==0) ? 0 : items_emb[idx]*8 + pos_w[t] + item_time[idx]
// writes both f32 (for E_recom) and f16 (WMMA A operand)
// ---------------------------------------------------------------------------
__global__ void k_build_seqs(const float* __restrict__ emb_ui, const float* __restrict__ item_time,
                             const float* __restrict__ pos_w, const int* __restrict__ log_seqs,
                             float* __restrict__ seqs, _Float16* __restrict__ seqs_h) {
  int i = blockIdx.x * blockDim.x + threadIdx.x;
  if (i >= BSZ * TLEN * HID) return;
  int d = i & 63;
  int bt = i >> 6;
  int t = bt % TLEN;
  int idx = log_seqs[bt];
  float v = 0.f;
  if (idx != 0)
    v = emb_ui[(size_t)(NU + idx) * HID + d] * 8.f + pos_w[t * HID + d]
        + item_time[(size_t)idx * HID + d];
  seqs[i] = v;
  seqs_h[i] = (_Float16)v;
}

// ---------------------------------------------------------------------------
// E_recom row per batch: layer_norm( mean_t seqs[b,t,:] )  (final_intent==1/T)
// ---------------------------------------------------------------------------
__global__ void k_erecom(const float* __restrict__ seqs, const float* __restrict__ ln_g,
                         const float* __restrict__ ln_b, float* __restrict__ erec) {
  int b = blockIdx.x;
  int d = threadIdx.x;           // 64 threads
  __shared__ float sm[HID];
  float x = 0.f;
  const float* sb = seqs + (size_t)b * TLEN * HID;
  for (int t = 0; t < TLEN; ++t) x += sb[t * HID + d];
  x *= (1.f / (float)TLEN);
  sm[d] = x;
  __syncthreads();
  float m = 0.f;
#pragma unroll
  for (int j = 0; j < HID; ++j) m += sm[j];
  m *= (1.f / HID);
  float v = 0.f;
#pragma unroll
  for (int j = 0; j < HID; ++j) { float dd = sm[j] - m; v += dd * dd; }
  v *= (1.f / HID);
  erec[b * HID + d] = (x - m) * rsqrtf(v + 1e-8f) * ln_g[d] + ln_b[d];
}

// ---------------------------------------------------------------------------
// GRU scan: one block per batch element; 192 threads hold whh rows in VGPRs,
// h lives in LDS. 200 sequential steps (latency-bound by construction).
// ---------------------------------------------------------------------------
__global__ void k_gru(const float* __restrict__ xp, const float* __restrict__ whh,
                      const float* __restrict__ bhh, float* __restrict__ Fu) {
  int b = blockIdx.x;
  int g = threadIdx.x;             // 0..191
  __shared__ float h[HID];
  __shared__ float hp[3 * HID];
  float w[HID];
#pragma unroll
  for (int i = 0; i < HID; ++i) w[i] = whh[(size_t)g * HID + i];
  float bg = bhh[g];
  if (g < HID) h[g] = 0.f;
  __syncthreads();
  const float* xpb = xp + (size_t)b * TLEN * (3 * HID);
  float* fub = Fu + (size_t)b * TLEN * HID;
  for (int t = 0; t < TLEN; ++t) {
    float acc = bg;
#pragma unroll
    for (int i = 0; i < HID; ++i) acc += w[i] * h[i];
    hp[g] = acc;
    __syncthreads();
    float hnew = 0.f;
    if (g < HID) {
      const float* xr = xpb + t * (3 * HID);
      float r = 1.f / (1.f + __expf(-(xr[g] + hp[g])));
      float z = 1.f / (1.f + __expf(-(xr[HID + g] + hp[HID + g])));
      float n = tanhf(xr[2 * HID + g] + r * hp[2 * HID + g]);
      hnew = (1.f - z) * n + z * h[g];
    }
    __syncthreads();
    if (g < HID) { h[g] = hnew; fub[t * HID + g] = hnew; }
    __syncthreads();
  }
}

// ---------------------------------------------------------------------------
// log_feats = erec[b] + layer_norm(Fu[b,t]);  pos/neg logits via item_emb_w.
// ---------------------------------------------------------------------------
__global__ void k_logits(const float* __restrict__ Fu, const float* __restrict__ erec,
                         const float* __restrict__ ln_g, const float* __restrict__ ln_b,
                         const float* __restrict__ item_w, const int* __restrict__ pos_seq,
                         const int* __restrict__ neg_seq, float* __restrict__ out) {
  int bt = (blockIdx.x * blockDim.x + threadIdx.x) >> 5;
  if (bt >= BSZ * TLEN) return;
  int lane = threadIdx.x & 31;
  int b = bt / TLEN;
  const float* fu = Fu + (size_t)bt * HID;
  float f0 = fu[lane], f1 = fu[lane + 32];
  float m = reduce_add_32(f0 + f1) * (1.f / HID);
  float d0 = f0 - m, d1 = f1 - m;
  float v = reduce_add_32(d0 * d0 + d1 * d1) * (1.f / HID);
  float rstd = rsqrtf(v + 1e-8f);
  float l0 = erec[b * HID + lane]      + d0 * rstd * ln_g[lane]      + ln_b[lane];
  float l1 = erec[b * HID + lane + 32] + d1 * rstd * ln_g[lane + 32] + ln_b[lane + 32];
  int ps = pos_seq[bt], ns = neg_seq[bt];
  const float* pw = item_w + (size_t)ps * HID;
  const float* nw = item_w + (size_t)ns * HID;
  float p = reduce_add_32(l0 * pw[lane] + l1 * pw[lane + 32]);
  float q = reduce_add_32(l0 * nw[lane] + l1 * nw[lane + 32]);
  if (lane == 0) {
    out[bt] = p;
    out[BSZ * TLEN + bt] = q;
  }
}

// ---------------------------------------------------------------------------
// contrast loss (deterministic coprime-stride stand-in for the jax permutes)
// ---------------------------------------------------------------------------
__global__ void k_contrast(const float* __restrict__ f, int fstride,
                           const float* __restrict__ g, const int* __restrict__ gidx,
                           int n, int pra, int prb, int pca, int pcb,
                           float scale, float* __restrict__ out) {
  int i = (blockIdx.x * blockDim.x + threadIdx.x) >> 5;
  if (i >= n) return;
  int lane = threadIdx.x & 31;
  int gi  = gidx ? gidx[i] : i;
  int pr  = (int)(((long long)i * pra + prb) % n);
  int gpi = gidx ? gidx[pr] : pr;
  const float* fr = f + (size_t)i * fstride;
  float pos = 0.f, neg = 0.f;
#pragma unroll
  for (int jj = 0; jj < 2; ++jj) {
    int j = lane + jj * 32;
    float fv = fr[j];
    pos += fv * g[(size_t)gi * HID + j];
    int pj = (j * pca + pcb) & 63;
    neg += fv * g[(size_t)gpi * HID + pj];
  }
  pos = reduce_add_32(pos);
  neg = reduce_add_32(neg);
  if (lane == 0) {
    pos *= (1.f / HID);
    neg *= (1.f / HID);
    float sp = 1.f / (1.f + __expf(-pos));
    float sn = 1.f / (1.f + __expf(-neg));
    float term = -logf(1e-8f + sp) - logf(1e-8f + (1.f - sn));
    atomicAdd(out, scale * term / (float)n);
  }
}

// ---------------------------------------------------------------------------
extern "C" void kernel_launch(void* const* d_in, const int* in_sizes, int n_in,
                              void* d_out, int out_size, void* d_ws, size_t ws_size,
                              hipStream_t stream) {
  const float* user_emb_w = (const float*)d_in[0];
  const float* item_emb_w = (const float*)d_in[1];
  const float* cate_emb_w = (const float*)d_in[2];
  const float* year_w  = (const float*)d_in[3];
  const float* month_w = (const float*)d_in[4];
  const float* day_w   = (const float*)d_in[5];
  const float* pos_w   = (const float*)d_in[6];
  // d_in[7] mu_w, d_in[8] sigma_w: dead (final_intent is uniform)
  const float* gc_w  = (const float*)d_in[9];
  const float* gc_b  = (const float*)d_in[10];
  const float* bi_w  = (const float*)d_in[11];
  const float* bi_b  = (const float*)d_in[12];
  const float* gcc_w = (const float*)d_in[13];
  const float* gcc_b = (const float*)d_in[14];
  const float* bic_w = (const float*)d_in[15];
  const float* bic_b = (const float*)d_in[16];
  const float* ln_g  = (const float*)d_in[17];
  const float* ln_b  = (const float*)d_in[18];
  const float* gru_wih = (const float*)d_in[19];
  const float* gru_whh = (const float*)d_in[20];
  const float* gru_bih = (const float*)d_in[21];
  const float* gru_bhh = (const float*)d_in[22];
  const float* ui_vals = (const float*)d_in[23];
  const float* uc_vals = (const float*)d_in[24];
  const float* it_vals = (const float*)d_in[25];
  // d_in[26] time_int: dead
  const int* user_ids = (const int*)d_in[27];
  const int* log_seqs = (const int*)d_in[28];
  // d_in[29..31] year/month/day: unused by forward
  const int* pos_seqs = (const int*)d_in[32];
  const int* neg_seqs = (const int*)d_in[33];
  const int* ui_rows = (const int*)d_in[34];
  const int* ui_cols = (const int*)d_in[35];
  const int* uc_rows = (const int*)d_in[36];
  const int* uc_cols = (const int*)d_in[37];
  const int* it_rows = (const int*)d_in[38];
  const int* it_cols = (const int*)d_in[39];
  float* out = (float*)d_out;

  // ---- workspace layout: f32 region, then f16 region ----------------------
  float* ws = (float*)d_ws;
  float* ego_ui    = ws;                         // 16000*64
  float* side_ui   = ego_ui  + N_UI * HID;
  float* acc_ui    = side_ui + N_UI * HID;
  float* ego_uc    = acc_ui  + N_UI * HID;       // 6400*64
  float* side_uc   = ego_uc  + N_UC * HID;
  float* acc_uc    = side_uc + N_UC * HID;
  float* times_emb = acc_uc  + N_UC * HID;       // 65*64
  float* item_time = times_emb + 65 * HID;       // 10000*64
  float* seqs      = item_time + NI * HID;       // 256*200*64
  float* erec      = seqs + BSZ * TLEN * HID;    // 256*64
  float* xp        = erec + BSZ * HID;           // 256*200*192
  float* Fu        = xp + BSZ * TLEN * 3 * HID;  // 256*200*64
  _Float16* hb     = (_Float16*)(Fu + BSZ * TLEN * HID);
  _Float16* side_h = hb;                         // N_UI*64 (reused for UC)
  _Float16* prod_h = side_h + N_UI * HID;
  _Float16* seqs_h = prod_h + N_UI * HID;        // 256*200*64
  _Float16* wgc_h  = seqs_h + BSZ * TLEN * HID;  // 3*64*64
  _Float16* wbi_h  = wgc_h + 3 * HID * HID;
  _Float16* wgcc_h = wbi_h + 3 * HID * HID;      // 2*64*64
  _Float16* wbic_h = wgcc_h + 2 * HID * HID;
  _Float16* wih_h  = wbic_h + 2 * HID * HID;     // 192*64

  // ---- one-shot weight conversions to f16 ---------------------------------
  k_cvt16<<<(3 * HID * HID + 255) / 256, 256, 0, stream>>>(gc_w, wgc_h, 3 * HID * HID);
  k_cvt16<<<(3 * HID * HID + 255) / 256, 256, 0, stream>>>(bi_w, wbi_h, 3 * HID * HID);
  k_cvt16<<<(2 * HID * HID + 255) / 256, 256, 0, stream>>>(gcc_w, wgcc_h, 2 * HID * HID);
  k_cvt16<<<(2 * HID * HID + 255) / 256, 256, 0, stream>>>(bic_w, wbic_h, 2 * HID * HID);
  k_cvt16<<<(3 * HID * HID + 255) / 256, 256, 0, stream>>>(gru_wih, wih_h, 3 * HID * HID);

  // --- NGCF user-item ------------------------------------------------------
  k_init2<<<(N_UI * HID + 255) / 256, 256, 0, stream>>>(
      user_emb_w, NU * HID, item_emb_w, ego_ui, acc_ui, N_UI * HID);
  for (int k = 0; k < 3; ++k) {
    k_zero<<<(N_UI * HID + 255) / 256, 256, 0, stream>>>(side_ui, N_UI * HID);
    k_spmm<<<1024, 256, 0, stream>>>(ui_rows, ui_cols, ui_vals, ego_ui, side_ui, NNZ_UI);
    k_prep_ab<<<(N_UI * HID + 255) / 256, 256, 0, stream>>>(
        side_ui, ego_ui, side_h, prod_h, N_UI * HID);
    k_ngcf_dense<<<N_UI / 64, 128, 0, stream>>>(
        side_h, prod_h, wgc_h + k * HID * HID, gc_b + k * HID,
        wbi_h + k * HID * HID, bi_b + k * HID, ego_ui, acc_ui, N_UI);
  }
  k_scale<<<(N_UI * HID + 255) / 256, 256, 0, stream>>>(acc_ui, N_UI * HID, 0.25f);

  // --- NGCF user-cate ------------------------------------------------------
  k_init2<<<(N_UC * HID + 255) / 256, 256, 0, stream>>>(
      user_emb_w, NU * HID, cate_emb_w, ego_uc, acc_uc, N_UC * HID);
  for (int k = 0; k < 2; ++k) {
    k_zero<<<(N_UC * HID + 255) / 256, 256, 0, stream>>>(side_uc, N_UC * HID);
    k_spmm<<<512, 256, 0, stream>>>(uc_rows, uc_cols, uc_vals, ego_uc, side_uc, NNZ_UC);
    k_prep_ab<<<(N_UC * HID + 255) / 256, 256, 0, stream>>>(
        side_uc, ego_uc, side_h, prod_h, N_UC * HID);
    k_ngcf_dense<<<N_UC / 64, 128, 0, stream>>>(
        side_h, prod_h, wgcc_h + k * HID * HID, gcc_b + k * HID,
        wbic_h + k * HID * HID, bic_b + k * HID, ego_uc, acc_uc, N_UC);
  }
  k_scale<<<(N_UC * HID + 255) / 256, 256, 0, stream>>>(acc_uc, N_UC * HID, 1.f / 3.f);

  // --- item_time = spmm(it, concat(year_w, month_w, day_w)) ----------------
  k_concat3<<<(65 * HID + 255) / 256, 256, 0, stream>>>(
      year_w, 20 * HID, month_w, 13 * HID, day_w, times_emb, 65 * HID);
  k_zero<<<(NI * HID + 255) / 256, 256, 0, stream>>>(item_time, NI * HID);
  k_spmm<<<256, 256, 0, stream>>>(it_rows, it_cols, it_vals, times_emb, item_time, NNZ_IT);

  // --- sequence features ---------------------------------------------------
  k_build_seqs<<<(BSZ * TLEN * HID + 255) / 256, 256, 0, stream>>>(
      acc_ui, item_time, pos_w, log_seqs, seqs, seqs_h);
  k_erecom<<<BSZ, HID, 0, stream>>>(seqs, ln_g, ln_b, erec);

  // --- GRU -----------------------------------------------------------------
  k_xp_gemm<<<(BSZ * TLEN) / 64, 128, 0, stream>>>(seqs_h, wih_h, gru_bih, xp);
  k_gru<<<BSZ, 3 * HID, 0, stream>>>(xp, gru_whh, gru_bhh, Fu);

  // --- logits --------------------------------------------------------------
  k_logits<<<(BSZ * TLEN) / 8, 256, 0, stream>>>(Fu, erec, ln_g, ln_b,
                                                 item_emb_w, pos_seqs, neg_seqs, out);

  // --- contrast losses (scalar at out[2*B*T]) ------------------------------
  k_zero<<<1, 32, 0, stream>>>(out + 2 * BSZ * TLEN, 1);
  k_contrast<<<(BSZ + 7) / 8, 256, 0, stream>>>(
      Fu + (TLEN - 1) * HID, TLEN * HID, acc_ui, user_ids, BSZ,
      77, 13, 29, 7, 0.1f, out + 2 * BSZ * TLEN);
  k_contrast<<<(NU + 7) / 8, 256, 0, stream>>>(
      acc_ui, HID, acc_uc, nullptr, NU,
      3001, 17, 29, 7, 0.1f, out + 2 * BSZ * TLEN);
}
